// PoseAttRouter_90022514524276
// MI455X (gfx1250) — matrix operand order repaired and checked by reading, not compiled, not run on hardware
//
#include <hip/hip_runtime.h>
#include <math.h>

typedef __attribute__((ext_vector_type(16))) _Float16 v16h;
typedef __attribute__((ext_vector_type(8)))  float    v8f;

#define BDIM 64
#define NDIM 128
#define CDIM 128
#define DDIM 32
#define EPSQ 1e-7f
#define TWO_OVER_PI 0.63661977236758134f

// ---------------------------------------------------------------------------
// Kernel 1: per-(b,c) thread. Stream A (coalesced over c) and Q (float4,
// coalesced), accumulate the weighted 4x4 outer-product matrix M, then do an
// in-register cyclic Jacobi eigensolve. Writes acts and K (= last ROW of the
// eigenvector matrix, columns sorted by ascending eigenvalue, faithfully
// replicating the reference's eigvect[:, -1, :] indexing).
// ---------------------------------------------------------------------------

#define JACOBI_ROT(p, q)                                                     \
  {                                                                          \
    float apq = m[p][q];                                                     \
    if (fabsf(apq) > 1e-30f) {                                               \
      float theta = 0.5f * (m[q][q] - m[p][p]) / apq;                        \
      float tt = copysignf(1.0f, theta) /                                    \
                 (fabsf(theta) + sqrtf(theta * theta + 1.0f));               \
      float cc = rsqrtf(tt * tt + 1.0f);                                     \
      float ss = tt * cc;                                                    \
      float app = m[p][p], aqq = m[q][q];                                    \
      m[p][p] = app - tt * apq;                                              \
      m[q][q] = aqq + tt * apq;                                              \
      m[p][q] = 0.0f;                                                        \
      m[q][p] = 0.0f;                                                        \
      _Pragma("unroll") for (int i = 0; i < 4; ++i) {                        \
        if (i != p && i != q) {                                              \
          float mip = m[i][p], miq = m[i][q];                                \
          m[i][p] = cc * mip - ss * miq;                                     \
          m[p][i] = m[i][p];                                                 \
          m[i][q] = ss * mip + cc * miq;                                     \
          m[q][i] = m[i][q];                                                 \
        }                                                                    \
      }                                                                      \
      _Pragma("unroll") for (int i = 0; i < 4; ++i) {                        \
        float vip = ev[i][p], viq = ev[i][q];                                \
        ev[i][p] = cc * vip - ss * viq;                                      \
        ev[i][q] = ss * vip + cc * viq;                                      \
      }                                                                      \
    }                                                                        \
  }

__global__ __launch_bounds__(256) void pose_eig_kernel(
    const float* __restrict__ A, const float* __restrict__ Q,
    float* __restrict__ acts, float* __restrict__ outq) {
  int t = blockIdx.x * blockDim.x + threadIdx.x;  // 0..8191
  int b = t >> 7;
  int c = t & (CDIM - 1);

  const float* Ap = A + (size_t)b * NDIM * CDIM + c;
  const float4* Q4 = (const float4*)Q;  // one float4 per (b,n,c)
  int qbase = b * NDIM * CDIM + c;

  float denom = 0.0f;
  float s00 = 0, s01 = 0, s02 = 0, s03 = 0, s11 = 0, s12 = 0, s13 = 0,
        s22 = 0, s23 = 0, s33 = 0;
  for (int n = 0; n < NDIM; ++n) {
    float a = Ap[(size_t)n * CDIM];
    float4 qv = Q4[qbase + n * CDIM];
    float q0 = qv.x + EPSQ, q1 = qv.y + EPSQ, q2 = qv.z + EPSQ,
          q3 = qv.w + EPSQ;
    denom += fabsf(a);
    s00 += a * q0 * q0;  s01 += a * q0 * q1;  s02 += a * q0 * q2;
    s03 += a * q0 * q3;  s11 += a * q1 * q1;  s12 += a * q1 * q2;
    s13 += a * q1 * q3;  s22 += a * q2 * q2;  s23 += a * q2 * q3;
    s33 += a * q3 * q3;
  }
  float inv = 1.0f / fmaxf(denom, 1e-12f);

  float m[4][4] = {{s00 * inv, s01 * inv, s02 * inv, s03 * inv},
                   {s01 * inv, s11 * inv, s12 * inv, s13 * inv},
                   {s02 * inv, s12 * inv, s22 * inv, s23 * inv},
                   {s03 * inv, s13 * inv, s23 * inv, s33 * inv}};
  float ev[4][4] = {{1, 0, 0, 0}, {0, 1, 0, 0}, {0, 0, 1, 0}, {0, 0, 0, 1}};

  for (int sw = 0; sw < 8; ++sw) {
    JACOBI_ROT(0, 1)
    JACOBI_ROT(0, 2)
    JACOBI_ROT(0, 3)
    JACOBI_ROT(1, 2)
    JACOBI_ROT(1, 3)
    JACOBI_ROT(2, 3)
  }

  float lam[4] = {m[0][0], m[1][1], m[2][2], m[3][3]};
  int idx[4] = {0, 1, 2, 3};
#define CSWAP(i, j)                                                          \
  {                                                                          \
    if (lam[i] > lam[j]) {                                                   \
      float tl = lam[i]; lam[i] = lam[j]; lam[j] = tl;                       \
      int ti = idx[i];   idx[i] = idx[j]; idx[j] = ti;                       \
    }                                                                        \
  }
  CSWAP(0, 1) CSWAP(2, 3) CSWAP(0, 2) CSWAP(1, 3) CSWAP(1, 2)

  acts[t] = lam[3] / (lam[0] + lam[1] + lam[2] + lam[3]);

  float row3[4] = {ev[3][0], ev[3][1], ev[3][2], ev[3][3]};
#pragma unroll
  for (int j = 0; j < 4; ++j) {
    int k = idx[j];
    float val = (k == 0) ? row3[0] : (k == 1) ? row3[1]
                : (k == 2) ? row3[2] : row3[3];
    outq[t * 4 + j] = val;
  }
}

// ---------------------------------------------------------------------------
// Kernel 2: one wave32 per (b,c). f32 softmax over N, then attention pooling
// out_f[b,c,:] = att(1x128) x V_bc(128x32) via V_WMMA_F32_16X16X32_F16:
// att -> A-matrix row 0 (16x32, contiguous k = 16h + s, same as B), V chunk -> B (32x16),
// f32 accumulate, read row 0 of D (VGPR0, lanes 0-15). 4 K-chunks x 2 d-halves.
// ---------------------------------------------------------------------------
__global__ __launch_bounds__(128) void pose_att_pool_kernel(
    const float* __restrict__ Q, const float* __restrict__ V,
    const float* __restrict__ Kq, float* __restrict__ outF) {
  __shared__ float att[4][NDIM];

  int lane = threadIdx.x & 31;
  int w = threadIdx.x >> 5;
  int bc = blockIdx.x * 4 + w;  // 0..8191
  int b = bc >> 7;
  int c = bc & (CDIM - 1);

  float k0 = Kq[bc * 4 + 0];
  float k1 = Kq[bc * 4 + 1];
  float k2 = Kq[bc * 4 + 2];
  float k3 = Kq[bc * 4 + 3];

  // ---- geodesic dist + softmax (f32, lane l handles n = l + 32*i) ----
  const float4* Q4 = (const float4*)Q;
  int qbase = b * NDIM * CDIM + c;
  float s[4];
#pragma unroll
  for (int i = 0; i < 4; ++i) {
    int n = lane + 32 * i;
    float4 qv = Q4[qbase + n * CDIM];
    float dot = qv.x * k0 + qv.y * k1 + qv.z * k2 + qv.w * k3;
    float ad = fminf(fabsf(dot), 1.0f - EPSQ);
    s[i] = -(acosf(ad) * TWO_OVER_PI);  // -dist
  }
  float mx = fmaxf(fmaxf(s[0], s[1]), fmaxf(s[2], s[3]));
#pragma unroll
  for (int off = 16; off > 0; off >>= 1)
    mx = fmaxf(mx, __shfl_xor(mx, off, 32));
  float e[4], sum = 0.0f;
#pragma unroll
  for (int i = 0; i < 4; ++i) {
    e[i] = __expf(s[i] - mx);
    sum += e[i];
  }
#pragma unroll
  for (int off = 16; off > 0; off >>= 1) sum += __shfl_xor(sum, off, 32);
  float scale = 1.0f / (sum * (float)NDIM);  // fold mean over N
#pragma unroll
  for (int i = 0; i < 4; ++i) att[w][lane + 32 * i] = e[i] * scale;
  __builtin_amdgcn_wave_barrier();  // order LDS store -> cross-lane LDS read

  // ---- WMMA pooling ----
  const float* Vp = V + (size_t)b * (NDIM * CDIM * DDIM) + (size_t)c * DDIM;
  int l16 = lane & 15;
  bool hi = lane >= 16;
  bool special = (lane == 0) || (lane == 16);
  float asc = special ? 1.0f : 0.0f;
  int abase_off = (lane == 16) ? 16 : 0;

  v8f acc0 = {};
  v8f acc1 = {};
#pragma unroll
  for (int k4 = 0; k4 < 4; ++k4) {
    int n0 = 32 * k4;
    // A fragment: only row M=0 nonzero. Lane 0: K 0-15; lane 16: K 16-31.
    int abase = n0 + abase_off;
    v16h a;
#pragma unroll
    for (int j = 0; j < 8; ++j) {
      a[j]     = (_Float16)(att[w][abase + j] * asc);
      a[8 + j] = (_Float16)(att[w][abase + 8 + j] * asc);
    }

    int rb = n0 + (hi ? 16 : 0);
    const float* vcol = Vp + l16;
    if (k4 < 3)  // prefetch next chunk's rows (global_prefetch_b8)
      __builtin_prefetch(vcol + (size_t)(rb + 32) * (CDIM * DDIM), 0, 0);

#pragma unroll
    for (int h = 0; h < 2; ++h) {
      v16h bf;
#pragma unroll
      for (int vv = 0; vv < 8; ++vv) {
        float f0 = vcol[(size_t)(rb + 2 * vv)     * (CDIM * DDIM) + 16 * h];
        float f1 = vcol[(size_t)(rb + 2 * vv + 1) * (CDIM * DDIM) + 16 * h];
        bf[2 * vv]     = (_Float16)f0;
        bf[2 * vv + 1] = (_Float16)f1;
      }
      if (h == 0)
        acc0 = __builtin_amdgcn_wmma_f32_16x16x32_f16(
            false, a, false, bf, (short)0, acc0, false, false);
      else
        acc1 = __builtin_amdgcn_wmma_f32_16x16x32_f16(
            false, a, false, bf, (short)0, acc1, false, false);
    }
  }

  // Row M=0 of D lives in VGPR0 of the accumulator, lanes 0-15 (N = lane).
  if (lane < 16) {
    outF[(size_t)bc * DDIM + lane]      = acc0[0];
    outF[(size_t)bc * DDIM + 16 + lane] = acc1[0];
  }
}

// ---------------------------------------------------------------------------
extern "C" void kernel_launch(void* const* d_in, const int* in_sizes, int n_in,
                              void* d_out, int out_size, void* d_ws,
                              size_t ws_size, hipStream_t stream) {
  (void)in_sizes; (void)n_in; (void)out_size; (void)d_ws; (void)ws_size;
  const float* A = (const float*)d_in[0];
  const float* Q = (const float*)d_in[1];
  const float* V = (const float*)d_in[2];

  float* acts = (float*)d_out;               // [B,C]    = 8192
  float* outq = acts + BDIM * CDIM;          // [B,C,4]  = 32768
  float* outf = outq + BDIM * CDIM * 4;      // [B,C,32] = 262144

  const int BC = BDIM * CDIM;  // 8192
  pose_eig_kernel<<<dim3(BC / 256), dim3(256), 0, stream>>>(A, Q, acts, outq);
  pose_att_pool_kernel<<<dim3(BC / 4), dim3(128), 0, stream>>>(Q, V, outq,
                                                               outf);
}
